// Model_Latent_Iterator_only_geometry_31396210933796
// MI455X (gfx1250) — compile-verified, matrix operand order checked by reading, not run on hardware
//
#include <hip/hip_runtime.h>
#include <hip/hip_bf16.h>

typedef __attribute__((ext_vector_type(16))) _Float16 v16h;
typedef __attribute__((ext_vector_type(8)))  float    v8f;

#define NPIX 1024
#define LFENCE() asm volatile("" ::: "memory")

struct KParams {
  const float *pix, *mask, *tnorm;
  const float *g1W,*g1b,*g2W,*g2b;
  const float *e1W,*e1b,*e2W,*e2b;
  const float *initLat;
  const float *l1W,*l1b,*l2W,*l2b;
  const float *lnLatG,*lnLatB,*lnMlatG,*lnMlatB,*lnMpixG,*lnMpixB,*lnPixG,*lnPixB;
  const float *gW,*gb,*m1W,*m1b,*m2W,*m2b;
  const float *outW;
  const float *p1W,*p1b,*p2W,*p2b;
  const float *kW,*kb,*oW,*ob,*qW,*qb,*vW,*vb;
  float *out;
};

// CDNA5 16-bit A-matrix 16x32 layout (ISA 7.12.2): lane L<16 holds row M=L,
// slots 0..7 -> K=0..7, slots 8..15 -> K=16..23; lane L+16 holds K=8..15 / 24..31.
__device__ __forceinline__ int kmap(int lane, int h) {
  return ((h & 8) << 1) | ((lane & 16) >> 1) | (h & 7);
}

// B (KxN) chunk from row-major global weight W[k*ncols + n]; mirrors A layout with N<->lanes.
__device__ __forceinline__ v16h loadB(const float* W, int ncols, int k0, int n0, int lane) {
  v16h b;
  int n = n0 + (lane & 15);
#pragma unroll
  for (int h = 0; h < 16; ++h) b[h] = (_Float16)W[(k0 + kmap(lane, h)) * ncols + n];
  return b;
}

// A 16x32 tile from LDS, row stride 32 floats; rowBase points at this lane's row.
__device__ __forceinline__ v16h loadA(const float* rowBase, int lane) {
  v16h a;
#pragma unroll
  for (int h = 0; h < 16; ++h) a[h] = (_Float16)rowBase[kmap(lane, h)];
  return a;
}

__device__ __forceinline__ v8f wmma16(v16h a, v16h b, v8f c) {
  return __builtin_amdgcn_wmma_f32_16x16x32_f16(false, a, false, b, (short)0, c, false, false);
}

__device__ __forceinline__ float wredSum(float v) {
#pragma unroll
  for (int off = 16; off > 0; off >>= 1) v += __shfl_xor(v, off, 32);
  return v;
}
__device__ __forceinline__ float wredMax(float v) {
#pragma unroll
  for (int off = 16; off > 0; off >>= 1) v = fmaxf(v, __shfl_xor(v, off, 32));
  return v;
}

__global__ __launch_bounds__(256, 1)
void perceiver_iter_kernel(KParams P) {
  __shared__ float sEmb[NPIX * 32];     // resident embedding, 128 KB
  __shared__ float sMu[NPIX];
  __shared__ float sRstd[NPIX];
  __shared__ float sUnion[4096];        // phase1: per-wave h-scratch (8x16x32); later: logits/attn (1024x4)
  __shared__ float sSpart[1024];        // per-wave S partials (8 x 4 x 32)
  __shared__ float sZ[32], sZn[32], sQ[32], sU[128];
  __shared__ float sAlpha[4], sFix[4], sHeadMax[4], sHeadSum[4];
  __shared__ float sRed[32];
  __shared__ float sS[128], sPooled[32], sComb[64], sT1[32], sH128[128], sGeo[64];

  const int tid  = threadIdx.x;
  const int lane = tid & 31;
  const int wave = tid >> 5;
  const int b    = blockIdx.x;
  const size_t pixBase = (size_t)b * NPIX;
  const int nlo = lane & 15;
  const int Mb  = (lane >> 4) << 3;     // D-layout: M = Mb + r, N = nlo

  if (tid < 32) sZ[tid] = P.initLat[tid];

  // ---- prologue: emb = init_emb2( relu(init_emb1(pix)) ) ----
#pragma unroll
  for (int pi = 0; pi < 4; ++pi) {
    int p = tid + pi * 256;
    float x0 = P.pix[(pixBase + p) * 2 + 0];
    float x1 = P.pix[(pixBase + p) * 2 + 1];
    for (int j = 0; j < 32; ++j) {
      float h = fmaf(x1, P.e1W[32 + j], fmaf(x0, P.e1W[j], P.e1b[j]));
      sEmb[p * 32 + j] = fmaxf(h, 0.f);
    }
  }
  __syncthreads();

  // ---- persistent weight tiles in B layout (f16) ----
  v16h Be2_0  = loadB(P.e2W, 32,  0,  0, lane), Be2_1  = loadB(P.e2W, 32,  0, 16, lane);
  v16h Bp1_00 = loadB(P.p1W, 32,  0,  0, lane), Bp1_01 = loadB(P.p1W, 32,  0, 16, lane);
  v16h Bp1_10 = loadB(P.p1W, 32, 32,  0, lane), Bp1_11 = loadB(P.p1W, 32, 32, 16, lane);
  v16h Bp2_0  = loadB(P.p2W, 32,  0,  0, lane), Bp2_1  = loadB(P.p2W, 32,  0, 16, lane);
  const float be2lo = P.e2b[nlo], be2hi = P.e2b[16 + nlo];
  const float bp1lo = P.p1b[nlo], bp1hi = P.p1b[16 + nlo];
  const float bp2lo = P.p2b[nlo], bp2hi = P.p2b[16 + nlo];

  // init_emb2 via WMMA (in-place over sEmb; A-loads precede stores, DS in-order per wave)
  for (int t = 0; t < 8; ++t) {
    int row0 = wave * 128 + t * 16;
    v16h a = loadA(&sEmb[(row0 + nlo) * 32], lane);
    v8f zc{};
    v8f c0 = wmma16(a, Be2_0, zc);
    v8f c1 = wmma16(a, Be2_1, zc);
    LFENCE();
#pragma unroll
    for (int r = 0; r < 8; ++r) {
      int row = row0 + Mb + r;
      sEmb[row * 32 + nlo]      = c0[r] + be2lo;
      sEmb[row * 32 + 16 + nlo] = c1[r] + be2hi;
    }
  }
  __syncthreads();

  for (int iter = 0; iter < 5; ++iter) {
    // A-chunk for the broadcast Z half of [emb | Z] (same for every row)
    v16h zA;
#pragma unroll
    for (int h = 0; h < 16; ++h) zA[h] = (_Float16)sZ[kmap(lane, h)];

    // ---- phase 1: emb += pix_upd2( relu(pix_upd1([emb|Z])) ) ----
    for (int t = 0; t < 8; ++t) {
      int row0 = wave * 128 + t * 16;
      v16h aE = loadA(&sEmb[(row0 + nlo) * 32], lane);
      v8f zc{};
      v8f c0 = wmma16(aE, Bp1_00, zc); c0 = wmma16(zA, Bp1_10, c0);
      v8f c1 = wmma16(aE, Bp1_01, zc); c1 = wmma16(zA, Bp1_11, c1);
      float* hs = &sUnion[wave * 512];
#pragma unroll
      for (int r = 0; r < 8; ++r) {
        hs[(Mb + r) * 32 + nlo]      = fmaxf(c0[r] + bp1lo, 0.f);
        hs[(Mb + r) * 32 + 16 + nlo] = fmaxf(c1[r] + bp1hi, 0.f);
      }
      LFENCE();
      v16h aH = loadA(&hs[nlo * 32], lane);
      v8f d0 = wmma16(aH, Bp2_0, zc);
      v8f d1 = wmma16(aH, Bp2_1, zc);
      LFENCE();
#pragma unroll
      for (int r = 0; r < 8; ++r) {
        int row = row0 + Mb + r;
        sEmb[row * 32 + nlo]      += d0[r] + bp2lo;
        sEmb[row * 32 + 16 + nlo] += d1[r] + bp2hi;
      }
    }
    __syncthreads();

    // ---- phase 2: per-pixel LN stats ----
#pragma unroll
    for (int pi = 0; pi < 4; ++pi) {
      int p = tid + pi * 256;
      float s = 0.f, s2 = 0.f;
      for (int c = 0; c < 32; ++c) { float v = sEmb[p * 32 + c]; s += v; s2 = fmaf(v, v, s2); }
      float mu  = s * (1.f / 32.f);
      float var = s2 * (1.f / 32.f) - mu * mu;
      sMu[p]   = mu;
      sRstd[p] = rsqrtf(var + 1e-5f);
    }
    __syncthreads();

    // ---- phase 3 (wave0): Zn, Q, U' = gamma * (Wk . Q/sqrt8), alpha, fix ----
    if (wave == 0) {
      float zv  = sZ[lane];
      float mu  = wredSum(zv) * (1.f / 32.f);
      float var = wredSum(zv * zv) * (1.f / 32.f) - mu * mu;
      float rs  = rsqrtf(var + 1e-5f);
      float zn  = (zv - mu) * rs * P.lnMlatG[lane] + P.lnMlatB[lane];
      sZn[lane] = zn;
      LFENCE();
      float q = P.qb[lane];
      for (int c = 0; c < 32; ++c) q = fmaf(sZn[c], P.qW[c * 32 + lane], q);
      q *= 0.35355339059f;  // 1/sqrt(HEAD_DIM)
      sQ[lane] = q;
      LFENCE();
      float g = P.lnMpixG[lane], bt = P.lnMpixB[lane];
#pragma unroll
      for (int h = 0; h < 4; ++h) {
        float u = 0.f;
#pragma unroll
        for (int d = 0; d < 8; ++d) u = fmaf(P.kW[lane * 32 + h * 8 + d], sQ[h * 8 + d], u);
        float up = g * u;
        sU[lane * 4 + h] = up;
        float alpha = wredSum(up);
        float gamma = wredSum(bt * u);
        if (lane == 0) {
          float beta = 0.f;
          for (int d = 0; d < 8; ++d) beta = fmaf(P.kb[h * 8 + d], sQ[h * 8 + d], beta);
          sAlpha[h] = alpha;
          sFix[h]   = gamma + beta;
        }
      }
    }
    __syncthreads();

    // ---- phase 4: logits = raw(E x U') fixed up with LN stats, via WMMA ----
    v16h bU;
#pragma unroll
    for (int h = 0; h < 16; ++h) {
      int k = kmap(lane, h);
      bU[h] = (nlo < 4) ? (_Float16)sU[k * 4 + nlo] : (_Float16)0.f;
    }
    for (int t = 0; t < 8; ++t) {
      int row0 = wave * 128 + t * 16;
      v16h aE = loadA(&sEmb[(row0 + nlo) * 32], lane);
      v8f zc{};
      v8f c = wmma16(aE, bU, zc);
      if (nlo < 4) {
#pragma unroll
        for (int r = 0; r < 8; ++r) {
          int p = row0 + Mb + r;
          float rs = sRstd[p];
          float lg = fmaf(rs, c[r], fmaf(-rs * sMu[p], sAlpha[nlo], sFix[nlo]));
          float mk = P.mask[pixBase + p];
          sUnion[p * 4 + nlo] = (mk == 0.f) ? -1e9f : lg;
        }
      }
    }
    __syncthreads();

    // ---- phase 5: per-head max over P ----
    {
      float mx[4] = {-3.0e38f, -3.0e38f, -3.0e38f, -3.0e38f};
#pragma unroll
      for (int pi = 0; pi < 4; ++pi) {
        int p = tid * 4 + pi;
#pragma unroll
        for (int h = 0; h < 4; ++h) mx[h] = fmaxf(mx[h], sUnion[p * 4 + h]);
      }
#pragma unroll
      for (int h = 0; h < 4; ++h) mx[h] = wredMax(mx[h]);
      if (lane == 0)
        for (int h = 0; h < 4; ++h) sRed[wave * 4 + h] = mx[h];
    }
    __syncthreads();
    if (tid < 4) {
      float m = -3.0e38f;
      for (int w = 0; w < 8; ++w) m = fmaxf(m, sRed[w * 4 + tid]);
      sHeadMax[tid] = m;
    }
    __syncthreads();

    // ---- phase 6: exp + per-head sum (attn left unnormalized in sUnion) ----
    {
      float sm[4] = {0.f, 0.f, 0.f, 0.f};
#pragma unroll
      for (int pi = 0; pi < 4; ++pi) {
        int p = tid * 4 + pi;
#pragma unroll
        for (int h = 0; h < 4; ++h) {
          float e = __expf(sUnion[p * 4 + h] - sHeadMax[h]);
          sUnion[p * 4 + h] = e;
          sm[h] += e;
        }
      }
#pragma unroll
      for (int h = 0; h < 4; ++h) sm[h] = wredSum(sm[h]);
      if (lane == 0)
        for (int h = 0; h < 4; ++h) sRed[wave * 4 + h] = sm[h];
    }
    __syncthreads();
    if (tid < 4) {
      float s = 0.f;
      for (int w = 0; w < 8; ++w) s += sRed[w * 4 + tid];
      sHeadSum[tid] = s;
    }
    __syncthreads();

    // ---- phase 7: S partials: acc_h[c] = sum_p attn*rstd*(E[p,c]-mu_p) ----
    {
      float inv0 = 1.f / sHeadSum[0], inv1 = 1.f / sHeadSum[1];
      float inv2 = 1.f / sHeadSum[2], inv3 = 1.f / sHeadSum[3];
      float a0 = 0.f, a1 = 0.f, a2 = 0.f, a3 = 0.f;
      for (int i = 0; i < 128; ++i) {
        int p = wave * 128 + i;
        float rs = sRstd[p];
        float ev = sEmb[p * 32 + lane] - sMu[p];
        a0 = fmaf(sUnion[p * 4 + 0] * inv0 * rs, ev, a0);
        a1 = fmaf(sUnion[p * 4 + 1] * inv1 * rs, ev, a1);
        a2 = fmaf(sUnion[p * 4 + 2] * inv2 * rs, ev, a2);
        a3 = fmaf(sUnion[p * 4 + 3] * inv3 * rs, ev, a3);
      }
      sSpart[wave * 128 +   0 + lane] = a0;
      sSpart[wave * 128 +  32 + lane] = a1;
      sSpart[wave * 128 +  64 + lane] = a2;
      sSpart[wave * 128 +  96 + lane] = a3;
    }
    __syncthreads();

    // ---- phase 8 (wave0): S, pooled, to_out, gated update, latent MLP ----
    if (wave == 0) {
      float g = P.lnMpixG[lane], bt = P.lnMpixB[lane];
#pragma unroll
      for (int h = 0; h < 4; ++h) {
        float T = 0.f;
        for (int w = 0; w < 8; ++w) T += sSpart[w * 128 + h * 32 + lane];
        sS[h * 32 + lane] = fmaf(g, T, bt);
      }
      LFENCE();
      {
        int h = lane >> 3;
        float pl = P.vb[lane];
        for (int c = 0; c < 32; ++c) pl = fmaf(sS[h * 32 + c], P.vW[c * 32 + lane], pl);
        sPooled[lane] = pl;
      }
      LFENCE();
      {
        float o = P.ob[lane];
        for (int c = 0; c < 32; ++c) o = fmaf(sPooled[c], P.oW[c * 32 + lane], o);
        sComb[lane]      = sZn[lane];
        sComb[32 + lane] = o;
      }
      LFENCE();
      {
        float t1 = P.m1b[lane];
        for (int c = 0; c < 64; ++c) t1 = fmaf(sComb[c], P.m1W[c * 32 + lane], t1);
        t1 = 0.5f * t1 * (1.f + erff(t1 * 0.70710678118f));  // exact gelu
        sT1[lane] = t1;
      }
      LFENCE();
      {
        float dz = P.m2b[lane];
        for (int c = 0; c < 32; ++c) dz = fmaf(sT1[c], P.m2W[c * 32 + lane], dz);
        float ga = P.gb[lane];
        for (int c = 0; c < 64; ++c) ga = fmaf(sComb[c], P.gW[c * 32 + lane], ga);
        ga = 1.f / (1.f + __expf(-ga));
        sZ[lane] = sZ[lane] + ga * dz;
      }
      LFENCE();
#pragma unroll
      for (int j = 0; j < 4; ++j) {
        int n = lane + 32 * j;
        float hh = P.l1b[n];
        for (int c = 0; c < 32; ++c) hh = fmaf(sZ[c], P.l1W[c * 128 + n], hh);
        sH128[n] = fmaxf(hh, 0.f);
      }
      LFENCE();
      {
        float acc = P.l2b[lane];
        for (int c = 0; c < 128; ++c) acc = fmaf(sH128[c], P.l2W[c * 32 + lane], acc);
        sZ[lane] += acc;
      }
    }
    __syncthreads();
  }

  // ---- epilogue: geometry head + time_norm passthrough ----
  if (wave == 0) {
#pragma unroll
    for (int j = 0; j < 2; ++j) {
      int n = lane + 32 * j;
      float a = P.g1b[n];
      for (int c = 0; c < 32; ++c) a = fmaf(sZ[c], P.g1W[c * 64 + n], a);
      sGeo[n] = fmaxf(a, 0.f);
    }
    LFENCE();
    if (lane < 3) {
      float a = P.g2b[lane];
      for (int c = 0; c < 64; ++c) a = fmaf(sGeo[c], P.g2W[c * 3 + lane], a);
      P.out[(size_t)b * 3 + lane] = a * P.outW[lane];
    }
    if (lane == 3) P.out[(size_t)gridDim.x * 3 + b] = P.tnorm[b];
  }
}

extern "C" void kernel_launch(void* const* d_in, const int* in_sizes, int n_in,
                              void* d_out, int out_size, void* d_ws, size_t ws_size,
                              hipStream_t stream) {
  (void)in_sizes; (void)n_in; (void)d_ws; (void)ws_size; (void)out_size;
  KParams P;
  // setup_inputs order: pix_data, mask, time_norm, then params leaves in sorted-key pytree order
  P.pix    = (const float*)d_in[0];
  P.mask   = (const float*)d_in[1];
  P.tnorm  = (const float*)d_in[2];
  P.g1W    = (const float*)d_in[3];  P.g1b   = (const float*)d_in[4];
  P.g2W    = (const float*)d_in[5];  P.g2b   = (const float*)d_in[6];
  P.e1W    = (const float*)d_in[7];  P.e1b   = (const float*)d_in[8];
  P.e2W    = (const float*)d_in[9];  P.e2b   = (const float*)d_in[10];
  P.initLat= (const float*)d_in[11];
  P.l1W    = (const float*)d_in[12]; P.l1b   = (const float*)d_in[13];
  P.l2W    = (const float*)d_in[14]; P.l2b   = (const float*)d_in[15];
  P.lnLatG = (const float*)d_in[16]; P.lnLatB= (const float*)d_in[17];
  P.lnMlatG= (const float*)d_in[18]; P.lnMlatB=(const float*)d_in[19];
  P.lnMpixG= (const float*)d_in[20]; P.lnMpixB=(const float*)d_in[21];
  P.lnPixG = (const float*)d_in[22]; P.lnPixB= (const float*)d_in[23];
  P.gW     = (const float*)d_in[24]; P.gb    = (const float*)d_in[25];
  P.m1W    = (const float*)d_in[26]; P.m1b   = (const float*)d_in[27];
  P.m2W    = (const float*)d_in[28]; P.m2b   = (const float*)d_in[29];
  P.outW   = (const float*)d_in[30];
  P.p1W    = (const float*)d_in[31]; P.p1b   = (const float*)d_in[32];
  P.p2W    = (const float*)d_in[33]; P.p2b   = (const float*)d_in[34];
  P.kW     = (const float*)d_in[35]; P.kb    = (const float*)d_in[36];
  P.oW     = (const float*)d_in[37]; P.ob    = (const float*)d_in[38];
  P.qW     = (const float*)d_in[39]; P.qb    = (const float*)d_in[40];
  P.vW     = (const float*)d_in[41]; P.vb    = (const float*)d_in[42];
  P.out    = (float*)d_out;

  perceiver_iter_kernel<<<4096, 256, 0, stream>>>(P);
}
